// LocalAttention_21182778704716
// MI455X (gfx1250) — compile-verified, matrix operand order checked by reading, not run on hardware
//
#include <hip/hip_runtime.h>
#include <hip/hip_bf16.h>
#include <math.h>

#define B_DIM   2
#define C_DIM   256
#define NUMTOK  4096        // 64*64
#define HD      32
#define NHEADS  8

typedef __attribute__((ext_vector_type(16))) _Float16 v16h;
typedef __attribute__((ext_vector_type(8)))  _Float16 h8;
typedef __attribute__((ext_vector_type(8)))  float    v8f;

__device__ __forceinline__ v8f wmma_f16(v16h a, v16h b, v8f c) {
  return __builtin_amdgcn_wmma_f32_16x16x32_f16(
      /*neg_a=*/false, a, /*neg_b=*/false, b,
      /*c_mod=*/(short)0, c, /*reuse_a=*/false, /*reuse_b=*/false);
}

// Load a 16x32 f16 A/B fragment row for this lane.
// Per ISA 7.12.2: lane<16 holds K = {0..7, 16..23}; lane>=16 holds K = {8..15, 24..31}.
// rowptr points at this lane's matrix row (A: row M=lane%16; B: column N=lane%16).
__device__ __forceinline__ v16h load_frag_g(const _Float16* rowptr, int lo) {
  union { v16h v; h8 h[2]; } u;
  u.h[0] = *reinterpret_cast<const h8*>(rowptr + lo);
  u.h[1] = *reinterpret_cast<const h8*>(rowptr + 16 + lo);
  return u.v;
}

// ---------------------------------------------------------------- conversions
__global__ void cvt_f32_f16_kernel(const float* __restrict__ src,
                                   _Float16* __restrict__ dst, int n) {
  int i = blockIdx.x * blockDim.x + threadIdx.x;
  if (i < n) dst[i] = (_Float16)src[i];
}

// x: [B, C, num] f32  ->  xT: [B, num, C] f16 (token-major)
__global__ void transpose_cvt_x_kernel(const float* __restrict__ x,
                                       _Float16* __restrict__ xT) {
  int i = blockIdx.x * blockDim.x + threadIdx.x;
  if (i >= B_DIM * C_DIM * NUMTOK) return;
  int b   = i / (C_DIM * NUMTOK);
  int rem = i - b * (C_DIM * NUMTOK);
  int c   = rem / NUMTOK;
  int n   = rem - c * NUMTOK;
  xT[((size_t)(b * NUMTOK + n)) * C_DIM + c] = (_Float16)x[i];
}

// ---------------------------------------------------------------- projections
// Per batch:  P[o, n] = sum_c W[o,c] * X[c,n] + bias[o], then * scale.
// 2x2 register blocking: each wave computes a 32(o) x 32(n) tile
// (2 A frags x 2 B frags -> 4 WMMAs per 8 b128 loads).
// mode 0: store token-major [b, n, C]   (q, k)
// mode 2: store channel-major [b, C, n] (v; serves directly as V^T fragments)
__global__ void __launch_bounds__(128)
proj_kernel(const _Float16* __restrict__ xT,   // [B, num, C]
            const _Float16* __restrict__ Wh,   // [C, C] row-major [o, c]
            const float*    __restrict__ bias, // [C]
            _Float16*       __restrict__ outp,
            float scale, int mode) {
  const int lane = threadIdx.x & 31;
  const int wave = threadIdx.x >> 5;
  const int lo   = (lane < 16) ? 0 : 8;
  const int col  = lane & 15;
  const int half = lane >> 4;

  const int tflat  = blockIdx.x;                  // B * num/32 tiles
  const int b      = tflat / (NUMTOK / 32);
  const int n_base = (tflat % (NUMTOK / 32)) * 32;
  const int o_base = blockIdx.y * 128 + wave * 32;

  const _Float16* arow0 = Wh + (size_t)(o_base + col) * C_DIM;
  const _Float16* arow1 = arow0 + (size_t)16 * C_DIM;
  const _Float16* brow0 = xT + ((size_t)(b * NUMTOK + n_base + col)) * C_DIM;
  const _Float16* brow1 = brow0 + (size_t)16 * C_DIM;

  v8f acc00 = {}, acc01 = {}, acc10 = {}, acc11 = {};
#pragma unroll
  for (int kb = 0; kb < C_DIM; kb += 32) {
    v16h a0 = load_frag_g(arow0 + kb, lo);
    v16h a1 = load_frag_g(arow1 + kb, lo);
    v16h x0 = load_frag_g(brow0 + kb, lo);
    v16h x1 = load_frag_g(brow1 + kb, lo);
    acc00 = wmma_f16(a0, x0, acc00);
    acc01 = wmma_f16(a0, x1, acc01);
    acc10 = wmma_f16(a1, x0, acc10);
    acc11 = wmma_f16(a1, x1, acc11);
  }

#pragma unroll
  for (int i = 0; i < 2; ++i) {
#pragma unroll
    for (int j = 0; j < 2; ++j) {
      const v8f& acc = (i == 0) ? ((j == 0) ? acc00 : acc01)
                                : ((j == 0) ? acc10 : acc11);
      const int n = n_base + 16 * j + col;
#pragma unroll
      for (int r = 0; r < 8; ++r) {
        int o = o_base + 16 * i + r + 8 * half;   // C/D layout: M = vgpr + 8*(lane/16)
        _Float16 hv = (_Float16)((acc[r] + bias[o]) * scale);
        if (mode == 2)
          outp[((size_t)(b * C_DIM + o)) * NUMTOK + n] = hv;
        else
          outp[((size_t)(b * NUMTOK + n)) * C_DIM + o] = hv;
      }
    }
  }
}

// ---------------------------------------------------------------- attention
// Flash-attention: each wave owns a 16-query tile of one (b, head).
// S(16x32) = q(16x32) * k^T via 2 WMMAs (K = hd = 32), online softmax,
// P staged through this wave's private LDS slice to become an A fragment,
// O += P * V via 2 WMMAs.  LDS ops of one wave complete in order on CDNA5,
// so only compiler scheduling fences are needed (no s_barrier).
__global__ void __launch_bounds__(128)
attn_kernel(const _Float16* __restrict__ q,     // [B, num, C] token-major
            const _Float16* __restrict__ k,     // [B, num, C] token-major
            const _Float16* __restrict__ vT,    // [B, C, num] channel-major
            _Float16*       __restrict__ fuseT) // [B, num, C] token-major
{
  __shared__ __align__(16) _Float16 lds[4][16][40];   // per-wave 16x32 P tile (+pad)

  const int lane = threadIdx.x & 31;
  const int wave = threadIdx.x >> 5;
  const int lo   = (lane < 16) ? 0 : 8;
  const int col  = lane & 15;
  const int half = lane >> 4;

  const int bh = blockIdx.y;          // 16 batch-heads
  const int b  = bh >> 3;
  const int h  = bh & 7;
  const int m_base = blockIdx.x * 64 + wave * 16;

  const _Float16* qrow = q + ((size_t)(b * NUMTOK + m_base + col)) * C_DIM + h * HD;
  v16h qa = load_frag_g(qrow, lo);

  float mrow[8], lrow[8];
  v8f o0 = {}, o1 = {};
#pragma unroll
  for (int r = 0; r < 8; ++r) { mrow[r] = -1e30f; lrow[r] = 0.0f; }

  for (int kb = 0; kb < NUMTOK; kb += 32) {
    // B fragment for S: column N = key token, elements over d (k row is contiguous)
    const _Float16* kr0 = k + ((size_t)(b * NUMTOK + kb + col)) * C_DIM + h * HD;
    v16h kf0 = load_frag_g(kr0, lo);
    v16h kf1 = load_frag_g(kr0 + 16 * C_DIM, lo);

    v8f z0 = {}, z1 = {};
    v8f s0 = wmma_f16(qa, kf0, z0);   // keys kb .. kb+15
    v8f s1 = wmma_f16(qa, kf1, z1);   // keys kb+16 .. kb+31

    // online softmax: row r of this lane group = r + 8*half; 16 cols across half-wave
    float alpha[8];
#pragma unroll
    for (int r = 0; r < 8; ++r) {
      float t = fmaxf(s0[r], s1[r]);
      t = fmaxf(t, __shfl_xor(t, 1, 32));
      t = fmaxf(t, __shfl_xor(t, 2, 32));
      t = fmaxf(t, __shfl_xor(t, 4, 32));
      t = fmaxf(t, __shfl_xor(t, 8, 32));
      float mn = fmaxf(mrow[r], t);
      alpha[r] = __expf(mrow[r] - mn);
      mrow[r]  = mn;
      float p0 = __expf(s0[r] - mn);
      float p1 = __expf(s1[r] - mn);
      s0[r] = p0; s1[r] = p1;
      float rs = p0 + p1;
      rs += __shfl_xor(rs, 1, 32);
      rs += __shfl_xor(rs, 2, 32);
      rs += __shfl_xor(rs, 4, 32);
      rs += __shfl_xor(rs, 8, 32);
      lrow[r] = lrow[r] * alpha[r] + rs;
      o0[r] *= alpha[r];
      o1[r] *= alpha[r];
    }

    // stage P (16 rows x 32 keys) into this wave's LDS slice; same-wave DS ops
    // are processed in order by hardware -> only a scheduling fence is needed.
#pragma unroll
    for (int r = 0; r < 8; ++r) {
      int row = r + 8 * half;
      lds[wave][row][col]      = (_Float16)s0[r];
      lds[wave][row][16 + col] = (_Float16)s1[r];
    }
    __builtin_amdgcn_wave_barrier();   // keep DS stores before DS loads

    union { v16h v; h8 hh[2]; } pu;
    pu.hh[0] = *reinterpret_cast<const h8*>(&lds[wave][col][lo]);
    pu.hh[1] = *reinterpret_cast<const h8*>(&lds[wave][col][16 + lo]);
    __builtin_amdgcn_wave_barrier();   // keep DS loads before next-iter stores

    // B fragment for O: column N = d, elements over keys (vT row is contiguous)
    const _Float16* vr0 = vT + ((size_t)(b * C_DIM + h * HD + col)) * NUMTOK + kb;
    v16h vf0 = load_frag_g(vr0, lo);                       // d = col
    v16h vf1 = load_frag_g(vr0 + (size_t)16 * NUMTOK, lo); // d = 16 + col

    o0 = wmma_f16(pu.v, vf0, o0);
    o1 = wmma_f16(pu.v, vf1, o1);
  }

#pragma unroll
  for (int r = 0; r < 8; ++r) {
    int tok = m_base + r + 8 * half;
    float inv = 1.0f / lrow[r];
    size_t base = ((size_t)(b * NUMTOK + tok)) * C_DIM + h * HD;
    fuseT[base + col]      = (_Float16)(o0[r] * inv);
    fuseT[base + 16 + col] = (_Float16)(o1[r] * inv);
  }
}

// ---------------------------------------------------------------- fusion conv + mask blend
// hybrid[o,n] = sum_{c<512} Wf[o,c] * cat([x;fuse])[c,n] + bf[o]
// out = hybrid * m + x * (1 - m), m = half-pixel bilinear upsample of 32x32 mask.
// 2x2 register blocking like proj_kernel.
__global__ void __launch_bounds__(128)
fuse_kernel(const _Float16* __restrict__ xT,     // [B, num, C]
            const _Float16* __restrict__ fuseT,  // [B, num, C]
            const _Float16* __restrict__ Wfh,    // [C, 2C] row-major
            const float*    __restrict__ bf,     // [C]
            const float*    __restrict__ x_f32,  // [B, C, num]
            const float*    __restrict__ mask,   // [B, 1, 32, 32]
            float*          __restrict__ out)    // [B, C, num]
{
  const int lane = threadIdx.x & 31;
  const int wave = threadIdx.x >> 5;
  const int lo   = (lane < 16) ? 0 : 8;
  const int col  = lane & 15;
  const int half = lane >> 4;

  const int tflat  = blockIdx.x;
  const int b      = tflat / (NUMTOK / 32);
  const int n_base = (tflat % (NUMTOK / 32)) * 32;
  const int o_base = blockIdx.y * 128 + wave * 32;

  const _Float16* arow0 = Wfh + (size_t)(o_base + col) * (2 * C_DIM);
  const _Float16* arow1 = arow0 + (size_t)16 * (2 * C_DIM);
  const size_t tokrow0  = ((size_t)(b * NUMTOK + n_base + col)) * C_DIM;
  const size_t tokrow1  = tokrow0 + (size_t)16 * C_DIM;

  v8f acc00 = {}, acc01 = {}, acc10 = {}, acc11 = {};
#pragma unroll
  for (int kb = 0; kb < 2 * C_DIM; kb += 32) {
    const _Float16* base = (kb < C_DIM) ? xT : (fuseT - C_DIM);
    v16h a0 = load_frag_g(arow0 + kb, lo);
    v16h a1 = load_frag_g(arow1 + kb, lo);
    v16h x0 = load_frag_g(base + tokrow0 + kb, lo);
    v16h x1 = load_frag_g(base + tokrow1 + kb, lo);
    acc00 = wmma_f16(a0, x0, acc00);
    acc01 = wmma_f16(a0, x1, acc01);
    acc10 = wmma_f16(a1, x0, acc10);
    acc11 = wmma_f16(a1, x1, acc11);
  }

  // bilinear mask values for the two token columns of this lane
  float mval[2];
#pragma unroll
  for (int j = 0; j < 2; ++j) {
    const int n  = n_base + 16 * j + col;
    const int py = n >> 6, px = n & 63;
    float fy = py * 0.5f - 0.25f;
    float fx = px * 0.5f - 0.25f;
    float fy0 = floorf(fy), fx0 = floorf(fx);
    float wy = fy - fy0,    wx = fx - fx0;
    int iy0 = (int)fy0, ix0 = (int)fx0;
    int iy1 = min(iy0 + 1, 31), ix1 = min(ix0 + 1, 31);
    iy0 = max(iy0, 0); ix0 = max(ix0, 0);
    const float* mb = mask + (size_t)b * 32 * 32;
    float m00 = mb[iy0 * 32 + ix0], m01 = mb[iy0 * 32 + ix1];
    float m10 = mb[iy1 * 32 + ix0], m11 = mb[iy1 * 32 + ix1];
    mval[j] = (m00 * (1.f - wx) + m01 * wx) * (1.f - wy)
            + (m10 * (1.f - wx) + m11 * wx) * wy;
  }

#pragma unroll
  for (int i = 0; i < 2; ++i) {
#pragma unroll
    for (int j = 0; j < 2; ++j) {
      const v8f& acc = (i == 0) ? ((j == 0) ? acc00 : acc01)
                                : ((j == 0) ? acc10 : acc11);
      const int n = n_base + 16 * j + col;
#pragma unroll
      for (int r = 0; r < 8; ++r) {
        int o = o_base + 16 * i + r + 8 * half;
        size_t idx = ((size_t)(b * C_DIM + o)) * NUMTOK + n;
        float hyb = acc[r] + bf[o];
        out[idx] = hyb * mval[j] + x_f32[idx] * (1.0f - mval[j]);
      }
    }
  }
}

// ---------------------------------------------------------------- launch
extern "C" void kernel_launch(void* const* d_in, const int* in_sizes, int n_in,
                              void* d_out, int out_size, void* d_ws, size_t ws_size,
                              hipStream_t stream) {
  (void)in_sizes; (void)n_in; (void)out_size; (void)ws_size;

  const float* x    = (const float*)d_in[0];
  const float* mask = (const float*)d_in[1];
  const float* Wq   = (const float*)d_in[2];
  const float* bq   = (const float*)d_in[3];
  const float* Wk   = (const float*)d_in[4];
  const float* bk   = (const float*)d_in[5];
  const float* Wv   = (const float*)d_in[6];
  const float* bv   = (const float*)d_in[7];
  const float* Wf   = (const float*)d_in[8];
  const float* bf   = (const float*)d_in[9];
  float* out = (float*)d_out;

  // workspace layout (halfs)
  const size_t NX = (size_t)B_DIM * NUMTOK * C_DIM;   // 2M halfs each
  _Float16* ws    = (_Float16*)d_ws;
  _Float16* xT    = ws;
  _Float16* qb    = xT + NX;
  _Float16* kb    = qb + NX;
  _Float16* vT    = kb + NX;
  _Float16* fuT   = vT + NX;
  _Float16* Wq16  = fuT + NX;
  _Float16* Wk16  = Wq16 + (size_t)C_DIM * C_DIM;
  _Float16* Wv16  = Wk16 + (size_t)C_DIM * C_DIM;
  _Float16* Wf16  = Wv16 + (size_t)C_DIM * C_DIM;

  const int NXE = B_DIM * C_DIM * NUMTOK;
  transpose_cvt_x_kernel<<<(NXE + 255) / 256, 256, 0, stream>>>(x, xT);
  cvt_f32_f16_kernel<<<(C_DIM * C_DIM + 255) / 256, 256, 0, stream>>>(Wq, Wq16, C_DIM * C_DIM);
  cvt_f32_f16_kernel<<<(C_DIM * C_DIM + 255) / 256, 256, 0, stream>>>(Wk, Wk16, C_DIM * C_DIM);
  cvt_f32_f16_kernel<<<(C_DIM * C_DIM + 255) / 256, 256, 0, stream>>>(Wv, Wv16, C_DIM * C_DIM);
  cvt_f32_f16_kernel<<<(2 * C_DIM * C_DIM + 255) / 256, 256, 0, stream>>>(Wf, Wf16, 2 * C_DIM * C_DIM);

  dim3 gproj(B_DIM * NUMTOK / 32, C_DIM / 128);
  const float qscale = 0.17677669529663687f;   // 1/sqrt(32)
  proj_kernel<<<gproj, 128, 0, stream>>>(xT, Wq16, bq, qb, qscale, 0);
  proj_kernel<<<gproj, 128, 0, stream>>>(xT, Wk16, bk, kb, 1.0f, 0);
  proj_kernel<<<gproj, 128, 0, stream>>>(xT, Wv16, bv, vT, 1.0f, 2);

  attn_kernel<<<dim3(NUMTOK / 64, B_DIM * NHEADS), 128, 0, stream>>>(qb, kb, vT, fuT);

  fuse_kernel<<<gproj, 128, 0, stream>>>(xT, fuT, Wf16, bf, x, mask, out);
}
